// BaseBranch_6811818132502
// MI455X (gfx1250) — compile-verified, matrix operand order checked by reading, not run on hardware
//
#include <hip/hip_runtime.h>
#include <hip/hip_bf16.h>

// MI455X / gfx1250 implementation.
// Strategy: fold channel-permutation, rot90 and BN into per-branch weight sets,
// then run all 8 branches as implicit-GEMM 3x3 convs on the tensor core path
// (v_wmma_f32_16x16x32_f16), f16 activations NHWC staged through LDS,
// f32 accumulation, atomic-max fusion of the channel/branch max.
// Workspace required: ~37.2 MB.

typedef _Float16 v8h  __attribute__((ext_vector_type(8)));
typedef _Float16 v16h __attribute__((ext_vector_type(16)));
typedef float    v8f  __attribute__((ext_vector_type(8)));

#define IN_CH   64
#define BATCH   8
#define MAP     96
#define DOM_ODD 112   // 96 + 2*8 halo (receptive radius of stack = 5 <= 8)
#define PAD_OFF 8
#define BN_EPS  1e-5f

// workspace layout (bytes, all 256-aligned)
static const size_t OFF_XH   = 0;          // f16 x, NHWC: 8*96*96*64*2  = 9,437,184
static const size_t OFF_WTS  = 9437184;    // f16 weights: 8br*3l*9tap*64o*64c*2 = 1,769,472
static const size_t OFF_BIAS = 11206656;   // f32 bias: 8*3*64*4 = 6,144
static const size_t OFF_ACTA = 11212800;   // f16 act ping: 8*112*112*64*2 = 12,845,056
static const size_t OFF_ACTB = 24057856;   // f16 act pong: 12,845,056
static const size_t OFF_OMAX = 36902912;   // u32/f32 max map: 8*96*96*4 = 294,912

__global__ void k_zero(unsigned* __restrict__ p, int n) {
    int i = blockIdx.x * blockDim.x + threadIdx.x;
    if (i < n) p[i] = 0u;
}

// NCHW f32 -> NHWC f16 (channel contiguous per pixel: 128B rows for b128 traffic)
__global__ void k_cvt_input(const float* __restrict__ x, _Float16* __restrict__ xh) {
    int i = blockIdx.x * blockDim.x + threadIdx.x;       // pixel id
    if (i >= BATCH * MAP * MAP) return;
    int w = i % MAP, h = (i / MAP) % MAP, b = i / (MAP * MAP);
    const float* src = x + (size_t)b * IN_CH * MAP * MAP + (size_t)h * MAP + w;
    _Float16* dst = xh + (size_t)i * IN_CH;
#pragma unroll 4
    for (int c = 0; c < IN_CH; c++) dst[c] = (_Float16)src[(size_t)c * MAP * MAP];
}

// Build effective per-branch weights: BN-scale fold, tap rotation (rot90 fold),
// input-channel roll fold for layer 0. Layout: [br][layer][tap][o][c] f16.
__global__ void k_prep_w(const float* __restrict__ w1, const float* __restrict__ b1,
                         const float* __restrict__ w2, const float* __restrict__ b2,
                         const float* __restrict__ w3, const float* __restrict__ b3,
                         const float* __restrict__ g,  const float* __restrict__ be,
                         const float* __restrict__ mn, const float* __restrict__ vr,
                         _Float16* __restrict__ wts, float* __restrict__ bias) {
    int i = blockIdx.x * blockDim.x + threadIdx.x;
    if (i >= 8 * 3 * 64 * 64) return;
    int c = i & 63, o = (i >> 6) & 63;
    int layer = (i >> 12) % 3, br = i / (64 * 64 * 3);
    const float* W  = (layer == 0) ? w1 : (layer == 1) ? w2 : w3;
    const float* Bb = (layer == 0) ? b1 : (layer == 1) ? b2 : b3;
    float scale = g[layer * 64 + o] * rsqrtf(vr[layer * 64 + o] + BN_EPS);
    int csrc = (layer == 0) ? ((c + 8 * br) & 63) : c;
    int k = br & 3;  // rotation count
    _Float16* dst = wts + (size_t)(br * 3 + layer) * 9 * 4096 + (size_t)o * 64 + c;
    for (int tap = 0; tap < 9; tap++) {
        int uy = tap / 3 - 1, ux = tap % 3 - 1;
        for (int t = 0; t < k; t++) { int ny = -ux, nx = uy; uy = ny; ux = nx; }
        int sy = uy + 1, sx = ux + 1;
        float wv = W[(((size_t)o * 64 + csrc) * 3 + sy) * 3 + sx] * scale;
        dst[(size_t)tap * 4096] = (_Float16)wv;
    }
    if (c == 0)
        bias[(br * 3 + layer) * 64 + o] =
            scale * Bb[o] + be[layer * 64 + o] - mn[layer * 64 + o] * scale;
}

__device__ __forceinline__ v16h cat8(v8h lo, v8h hi) {
    return __builtin_shufflevector(lo, hi, 0, 1, 2, 3, 4, 5, 6, 7,
                                          8, 9, 10, 11, 12, 13, 14, 15);
}

// One fused conv+BN+ReLU layer as implicit GEMM on WMMA.
// in: f16 NHWC, domain in_dom, read with coordinate shift in_off (zero outside).
// out: f16 NHWC domain D (layers 0/1), or atomic-max into omax (layer 2).
__global__ __launch_bounds__(256)
void k_conv(const _Float16* __restrict__ in, int in_dom, int in_off,
            const _Float16* __restrict__ wts_l, const float* __restrict__ bias_l,
            _Float16* __restrict__ out, unsigned* __restrict__ omax,
            int D, int dil, int layer3, int out_off) {
    __shared__ _Float16 smem[20 * 20 * 64];   // max halo case (dil=2): 51,200 B

    const int tilesx = D >> 4;
    const int tY = blockIdx.x / tilesx, tX = blockIdx.x % tilesx;
    const int b  = blockIdx.y;
    const int Wl = 16 + 2 * dil;
    const int npix = Wl * Wl;
    const int oy0 = tY * 16, ox0 = tX * 16;

    // ---- stage input tile + halo into LDS (zero pad outside source domain) ----
    for (int p = threadIdx.x; p < npix; p += 256) {
        int ly = p / Wl, lx = p % Wl;
        int sy = oy0 - dil + ly - in_off;
        int sx = ox0 - dil + lx - in_off;
        uint4* dst = (uint4*)&smem[p * 64];
        if ((unsigned)sy < (unsigned)in_dom && (unsigned)sx < (unsigned)in_dom) {
            const uint4* src =
                (const uint4*)&in[(((size_t)b * in_dom + sy) * in_dom + sx) * 64];
#pragma unroll
            for (int q = 0; q < 8; q++) dst[q] = src[q];
        } else {
            uint4 z = {0u, 0u, 0u, 0u};
#pragma unroll
            for (int q = 0; q < 8; q++) dst[q] = z;
        }
    }
    __syncthreads();

    const int wave = threadIdx.x >> 5;
    const int lane = threadIdx.x & 31;
    const int lh = lane >> 4;    // lane half selects K sub-block (+8)
    const int lc = lane & 15;    // M (out-ch) for A, N (pixel col) for B
    const int r0 = wave * 2;     // this wave owns output rows r0, r0+1
    const int kb = lh * 8;

    v8f acc[4][2] = {};          // [out-ch group][row]

    for (int tap = 0; tap < 9; tap++) {
        const int dy = (tap / 3 - 1) * dil, dx = (tap % 3 - 1) * dil;
        const int px  = lc + dil + dx;
        const int py0 = r0 + dil + dy;
#pragma unroll
        for (int kc = 0; kc < 2; kc++) {
            const int c0 = kc * 32 + kb;
            // B fragments: activation column per lane, K-striped per ISA layout
            const _Float16* s0 = &smem[(py0 * Wl + px) * 64 + c0];
            const _Float16* s1 = &smem[((py0 + 1) * Wl + px) * 64 + c0];
            v16h bf0 = cat8(*(const v8h*)s0, *(const v8h*)(s0 + 16));
            v16h bf1 = cat8(*(const v8h*)s1, *(const v8h*)(s1 + 16));
#pragma unroll
            for (int oc = 0; oc < 4; oc++) {
                // A fragment: weight row per lane (pre-packed [tap][o][c])
                const _Float16* aw =
                    &wts_l[(size_t)tap * 4096 + (size_t)(oc * 16 + lc) * 64 + c0];
                v16h af = cat8(*(const v8h*)aw, *(const v8h*)(aw + 16));
                acc[oc][0] = __builtin_amdgcn_wmma_f32_16x16x32_f16(
                    false, af, false, bf0, (short)0, acc[oc][0], false, false);
                acc[oc][1] = __builtin_amdgcn_wmma_f32_16x16x32_f16(
                    false, af, false, bf1, (short)0, acc[oc][1], false, false);
            }
        }
    }

    // ---- epilogue: bias + ReLU; store f16 NHWC, or fuse channel max ----
#pragma unroll
    for (int ri = 0; ri < 2; ri++) {
        const int oy = oy0 + r0 + ri;
        const int ox = ox0 + lc;
        if (layer3) {
            float m = 0.0f;  // ReLU lower bound == init, so max == max(relu(.))
#pragma unroll
            for (int oc = 0; oc < 4; oc++) {
                const int chb = oc * 16 + lh * 8;
#pragma unroll
                for (int r = 0; r < 8; r++)
                    m = fmaxf(m, acc[oc][ri][r] + bias_l[chb + r]);
            }
            int py = oy - out_off, pxo = ox - out_off;
            if ((unsigned)py < (unsigned)MAP && (unsigned)pxo < (unsigned)MAP)
                atomicMax(&omax[((size_t)b * MAP + py) * MAP + pxo],
                          __float_as_uint(m));   // valid: all candidates >= 0
        } else {
#pragma unroll
            for (int oc = 0; oc < 4; oc++) {
                const int chb = oc * 16 + lh * 8;
                v8h ov;
#pragma unroll
                for (int r = 0; r < 8; r++)
                    ov[r] = (_Float16)fmaxf(acc[oc][ri][r] + bias_l[chb + r], 0.0f);
                *(v8h*)&out[(((size_t)b * D + oy) * D + ox) * 64 + chb] = ov;
            }
        }
    }
}

__global__ void k_final(const unsigned* __restrict__ omax, float* __restrict__ out, int n) {
    int i = blockIdx.x * blockDim.x + threadIdx.x;
    if (i >= n) return;
    float v = __uint_as_float(omax[i]);
    float s = 1.0f / (1.0f + __expf(-v));
    out[i] = fminf(fmaxf(s, 1.0e-4f), 1.0f - 1.0e-4f);
}

extern "C" void kernel_launch(void* const* d_in, const int* in_sizes, int n_in,
                              void* d_out, int out_size, void* d_ws, size_t ws_size,
                              hipStream_t stream) {
    const float* x    = (const float*)d_in[0];
    // d_in[1] (perms) folded analytically into layer-0 weights.
    const float* dcnw = (const float*)d_in[2];
    const float* dcnb = (const float*)d_in[3];
    const float* c2w  = (const float*)d_in[4];
    const float* c2b  = (const float*)d_in[5];
    const float* c3w  = (const float*)d_in[6];
    const float* c3b  = (const float*)d_in[7];
    const float* bng  = (const float*)d_in[8];
    const float* bnb  = (const float*)d_in[9];
    const float* bnm  = (const float*)d_in[10];
    const float* bnv  = (const float*)d_in[11];

    char* ws = (char*)d_ws;
    _Float16* xh   = (_Float16*)(ws + OFF_XH);
    _Float16* wts  = (_Float16*)(ws + OFF_WTS);
    float*    bias = (float*)   (ws + OFF_BIAS);
    _Float16* actA = (_Float16*)(ws + OFF_ACTA);
    _Float16* actB = (_Float16*)(ws + OFF_ACTB);
    unsigned* omax = (unsigned*)(ws + OFF_OMAX);

    const int npx = BATCH * MAP * MAP;
    k_zero<<<(npx + 255) / 256, 256, 0, stream>>>(omax, npx);
    k_cvt_input<<<(npx + 255) / 256, 256, 0, stream>>>(x, xh);
    k_prep_w<<<(8 * 3 * 64 * 64) / 256, 256, 0, stream>>>(
        dcnw, dcnb, c2w, c2b, c3w, c3b, bng, bnb, bnm, bnv, wts, bias);

    for (int br = 0; br < 8; br++) {
        const bool odd = (br & 1) != 0;
        const int D   = odd ? DOM_ODD : MAP;
        const int off = odd ? PAD_OFF : 0;
        dim3 grid((D / 16) * (D / 16), BATCH);

        const _Float16* wl0 = wts + (size_t)(br * 3 + 0) * 9 * 4096;
        const _Float16* wl1 = wts + (size_t)(br * 3 + 1) * 9 * 4096;
        const _Float16* wl2 = wts + (size_t)(br * 3 + 2) * 9 * 4096;
        const float* bl0 = bias + (br * 3 + 0) * 64;
        const float* bl1 = bias + (br * 3 + 1) * 64;
        const float* bl2 = bias + (br * 3 + 2) * 64;

        // layer 0: dil 1, reads x (domain 96) with zero halo shift for odd branches
        k_conv<<<grid, 256, 0, stream>>>(xh,   MAP, off, wl0, bl0, actA, omax, D, 1, 0, 0);
        // layer 1: dil 2
        k_conv<<<grid, 256, 0, stream>>>(actA, D,   0,   wl1, bl1, actB, omax, D, 2, 0, 0);
        // layer 2: dil 2, fused channel/branch max into omax (crop via out_off)
        k_conv<<<grid, 256, 0, stream>>>(actB, D,   0,   wl2, bl2, actA, omax, D, 2, 1, off);
    }
    k_final<<<(npx + 255) / 256, 256, 0, stream>>>(omax, (float*)d_out, npx);
}
